// ImageMoE_34574486732891
// MI455X (gfx1250) — matrix operand order repaired and checked by reading, not compiled, not run on hardware
//
#include <hip/hip_runtime.h>

// ============================================================================
// ViT-MoE forward for MI455X (gfx1250, wave32).
// Heavy GEMMs (QKV / attn-proj / 2x MoE expert FFN, ~352 GFLOP) run on
// v_wmma_f32_16x16x32_bf16: 128x128 block tile, 8 waves x (32x64) wave tile,
// K-step 32, double-buffered LDS. Weights are pre-converted ONCE per launch to
// bf16 + transposed [N,K], so the GEMM hot loop is pure b128 traffic + WMMA
// (no conversion VALU, no scattered DS stores). Templated branch-free
// epilogues fuse bias/residual/ReLU/router-gate scaling.
// ============================================================================

#define T_TOK 1024   // B*S = 16*64 tokens
#define DMODEL 1024
#define DFF 4096
#define NEXP 10
#define HS 128       // head size
#define NH 8

typedef __attribute__((ext_vector_type(16))) __bf16 v16bf;
typedef __attribute__((ext_vector_type(8)))  float  v8f;

__device__ __forceinline__ unsigned short f2bf(float f) {
  union { float f; unsigned u; } v; v.f = f;
  unsigned r = v.u + 0x7FFFu + ((v.u >> 16) & 1u);   // round-to-nearest-even
  return (unsigned short)(r >> 16);
}
__device__ __forceinline__ float bf2f(unsigned short h) {
  union { unsigned u; float f; } v; v.u = ((unsigned)h) << 16;
  return v.f;
}

// ---------------------------------------------------------------------------
// One-time weight prep: W[K,N] f32  ->  Wt[N,K] bf16  (32x32 tiles via LDS).
// grid.z = expert index.
// ---------------------------------------------------------------------------
__global__ __launch_bounds__(256) void k_transpose_bf16(
    const float* __restrict__ W, long long strideWe,
    unsigned short* __restrict__ Wt, long long strideWte, int K, int N) {
  W  += (long long)blockIdx.z * strideWe;
  Wt += (long long)blockIdx.z * strideWte;
  const int n0 = blockIdx.x * 32, k0 = blockIdx.y * 32;
  const int tid = threadIdx.x;
  __shared__ unsigned short tile[32][34];   // [k][n], padded
  const int c = tid & 31, r8 = tid >> 5;
  #pragma unroll
  for (int it = 0; it < 4; ++it) {
    int r = r8 + it * 8;
    tile[r][c] = f2bf(W[(size_t)(k0 + r) * N + n0 + c]);
  }
  __syncthreads();
  const int c2 = (tid & 15) * 2, rr = tid >> 4;
  #pragma unroll
  for (int it = 0; it < 2; ++it) {
    int n = rr + it * 16;
    unsigned lo = tile[c2][n], hi = tile[c2 + 1][n];
    *(unsigned*)&Wt[(size_t)(n0 + n) * K + k0 + c2] = lo | (hi << 16);
  }
}

// ---------------------------------------------------------------------------
// Patchify + patch embedding: h0[t,d] = sum_p patch[t,p] * Wpe[p,d] + bpe[d]
// ---------------------------------------------------------------------------
__global__ __launch_bounds__(256) void k_patch_embed(
    const float* __restrict__ x, const float* __restrict__ Wpe,
    const float* __restrict__ bpe, float* __restrict__ h0) {
  int t = blockIdx.x, tid = threadIdx.x;
  __shared__ float pv[48];
  if (tid < 48) {
    int b = t >> 6, n = t & 63;
    int hp = n >> 3, wp = n & 7;
    int c = tid >> 4, rem = tid & 15, pr = rem >> 2, pc = rem & 3;
    pv[tid] = x[((size_t)(b * 3 + c) * 32 + (hp * 4 + pr)) * 32 + (wp * 4 + pc)];
  }
  __syncthreads();
  for (int d = tid; d < DMODEL; d += 256) {
    float acc = bpe[d];
    #pragma unroll
    for (int p = 0; p < 48; ++p) acc += pv[p] * Wpe[p * DMODEL + d];
    h0[(size_t)t * DMODEL + d] = acc;
  }
}

// ---------------------------------------------------------------------------
// LayerNorm (f32 in) -> bf16 out.  One block per token.
// ---------------------------------------------------------------------------
__global__ __launch_bounds__(256) void k_layernorm(
    const float* __restrict__ x, const float* __restrict__ g,
    const float* __restrict__ b, unsigned short* __restrict__ y) {
  int t = blockIdx.x, tid = threadIdx.x;
  const float* xr = x + (size_t)t * DMODEL;
  __shared__ float s1[256], s2[256];
  float a = 0.f, a2 = 0.f;
  for (int d = tid; d < DMODEL; d += 256) { float v = xr[d]; a += v; a2 += v * v; }
  s1[tid] = a; s2[tid] = a2; __syncthreads();
  for (int s = 128; s > 0; s >>= 1) {
    if (tid < s) { s1[tid] += s1[tid + s]; s2[tid] += s2[tid + s]; }
    __syncthreads();
  }
  float mean = s1[0] * (1.f / DMODEL);
  float var = s2[0] * (1.f / DMODEL) - mean * mean;
  float rstd = rsqrtf(var + 1e-5f);
  for (int d = tid; d < DMODEL; d += 256)
    y[(size_t)t * DMODEL + d] = f2bf((xr[d] - mean) * rstd * g[d] + b[d]);
}

// ---------------------------------------------------------------------------
// Tiled WMMA GEMM: C[M,N] = f(A_bf16[M,K] @ Bt_bf16[N,K]^T + bias (+res)),
// optional per-row gate scale; store f32 and/or bf16.
// Block 128x128, 8 waves (4x2), wave tile 32x64 (2x4 subtiles), K-step 32,
// double-buffered LDS.  grid.z = expert.  FLAGS = compile-time epilogue.
// ---------------------------------------------------------------------------
enum { F_BIAS = 1, F_RES = 2, F_RELU = 4, F_SCALE = 8, F_OUTF = 16, F_OUTB = 32 };

template <int FLAGS>
__global__ __launch_bounds__(256) void k_wmma_gemm(
    const unsigned short* __restrict__ A, long long strideAe,
    const unsigned short* __restrict__ Bt, long long strideBe,
    const float* __restrict__ bias, long long strideBiasE,
    float* Cf, long long strideCfE,
    unsigned short* Cb, long long strideCbE,
    const float* __restrict__ residual,
    const float* __restrict__ rowscale, int rsStride, long long strideRsE,
    int M, int N, int K) {
  const int e = blockIdx.z;
  A  += (long long)e * strideAe;
  Bt += (long long)e * strideBe;
  if (FLAGS & F_BIAS) bias += (long long)e * strideBiasE;
  if (FLAGS & F_OUTF) Cf += (long long)e * strideCfE;
  if (FLAGS & F_OUTB) Cb += (long long)e * strideCbE;
  if (FLAGS & F_SCALE) rowscale += (long long)e * strideRsE;

  const int n0 = blockIdx.x * 128, m0 = blockIdx.y * 128;
  const int tid = threadIdx.x, lane = tid & 31, wave = tid >> 5;
  const int wm = wave >> 1, wn = wave & 1;      // 4 x 2 wave grid (32x64 tiles)
  const int l15 = lane & 15;
  const int khalf = (lane >> 4) * 8;            // A frag K-offset (ISA layout)
  const int kb16  = (lane >> 4) * 16;           // unused; B frags also use khalf

  // Double-buffered LDS: A 128x32 bf16 and Bt 128x32 bf16, row stride 40.
  __shared__ unsigned short As[2 * 128 * 40];
  __shared__ unsigned short Bs[2 * 128 * 40];

  const v8f vz = {0.f, 0.f, 0.f, 0.f, 0.f, 0.f, 0.f, 0.f};
  v8f acc[2][4];
  #pragma unroll
  for (int i = 0; i < 2; ++i)
    #pragma unroll
    for (int j = 0; j < 4; ++j) acc[i][j] = vz;

  // Identical staging for A (m-rows) and Bt (n-rows): 256 thr * 16 halves.
  const int srow = tid >> 1, scol = (tid & 1) * 16;
  const unsigned short* Ag  = A  + (size_t)(m0 + srow) * K + scol;
  const unsigned short* Btg = Bt + (size_t)(n0 + srow) * K + scol;

  const int KT = K / 32;
  uint4 ar0, ar1, br0, br1;

  auto load_regs = [&](int kt) {
    ar0 = *(const uint4*)(Ag + kt * 32);
    ar1 = *(const uint4*)(Ag + kt * 32 + 8);
    br0 = *(const uint4*)(Btg + kt * 32);
    br1 = *(const uint4*)(Btg + kt * 32 + 8);
  };
  auto store_tile = [&](int buf) {
    unsigned short* as = As + buf * (128 * 40) + srow * 40 + scol;
    *(uint4*)(as) = ar0;
    *(uint4*)(as + 8) = ar1;
    unsigned short* bs = Bs + buf * (128 * 40) + srow * 40 + scol;
    *(uint4*)(bs) = br0;
    *(uint4*)(bs + 8) = br1;
  };

  load_regs(0);
  store_tile(0);
  __syncthreads();

  for (int kt = 0; kt < KT; ++kt) {
    const int cur = kt & 1;
    const bool more = (kt + 1) < KT;
    if (more) load_regs(kt + 1);
    if (kt + 2 < KT) {
      __builtin_prefetch(Ag + (kt + 2) * 32, 0, 1);
      __builtin_prefetch(Btg + (kt + 2) * 32, 0, 1);
    }

    union { uint4 q[2]; v16bf v; } af[2], bf[4];
    #pragma unroll
    for (int i = 0; i < 2; ++i) {
      const unsigned short* ab =
          As + cur * (128 * 40) + (wm * 32 + i * 16 + l15) * 40;
      af[i].q[0] = *(const uint4*)(ab + khalf);        // K 0..7 (+khalf)
      af[i].q[1] = *(const uint4*)(ab + 16 + khalf);   // K 16..23 (+khalf)
    }
    #pragma unroll
    for (int j = 0; j < 4; ++j) {
      const unsigned short* bb =
          Bs + cur * (128 * 40) + (wn * 64 + j * 16 + l15) * 40;
      bf[j].q[0] = *(const uint4*)(bb + khalf);        // K 0..7 (+khalf)
      bf[j].q[1] = *(const uint4*)(bb + 16 + khalf);   // K 16..23 (+khalf)
    }
    #pragma unroll
    for (int i = 0; i < 2; ++i)
      #pragma unroll
      for (int j = 0; j < 4; ++j)
        acc[i][j] = __builtin_amdgcn_wmma_f32_16x16x32_bf16(
            false, af[i].v, false, bf[j].v, (short)0, acc[i][j], false, false);

    if (more) store_tile((kt + 1) & 1);
    __syncthreads();
  }

  // Epilogue. C 16x16 layout: VGPR r -> M = r + (lane>=16)*8, N = lane&15.
  #pragma unroll
  for (int i = 0; i < 2; ++i) {
    #pragma unroll
    for (int r = 0; r < 8; ++r) {
      const int row = m0 + wm * 32 + i * 16 + (lane >> 4) * 8 + r;
      float sc = 1.f;
      if (FLAGS & F_SCALE) sc = rowscale[(size_t)row * rsStride];
      #pragma unroll
      for (int j = 0; j < 4; ++j) {
        const int cn = n0 + wn * 64 + j * 16 + l15;
        float val = acc[i][j][r];
        if (FLAGS & F_BIAS) val += bias[cn];
        if (FLAGS & F_RES) val += residual[(size_t)row * N + cn];
        if (FLAGS & F_RELU) val = fmaxf(val, 0.f);
        if (FLAGS & F_SCALE) val *= sc;
        const size_t idx = (size_t)row * N + cn;
        if (FLAGS & F_OUTF) Cf[idx] = val;
        if (FLAGS & F_OUTB) Cb[idx] = f2bf(val);
      }
    }
  }
}

// ---------------------------------------------------------------------------
// Fused attention core per (b, h, q): scores + causal softmax + A@V.
// ---------------------------------------------------------------------------
__global__ __launch_bounds__(128) void k_attention(
    const unsigned short* __restrict__ q, const unsigned short* __restrict__ k,
    const unsigned short* __restrict__ v, unsigned short* __restrict__ out) {
  int qi = blockIdx.x, h = blockIdx.y, b = blockIdx.z;
  int d = threadIdx.x;
  int t = b * 64 + qi;
  __shared__ float qrow[HS];
  __shared__ float sc[64];
  __shared__ float inv;
  qrow[d] = bf2f(q[(size_t)t * DMODEL + h * HS + d]);
  __syncthreads();
  if (d < 64) {
    float s;
    if (d <= qi) {
      s = 0.f;
      const unsigned short* kr = k + (size_t)(b * 64 + d) * DMODEL + h * HS;
      #pragma unroll 8
      for (int j = 0; j < HS; ++j) s += qrow[j] * bf2f(kr[j]);
      s *= 0.08838834764831845f;  // 1/sqrt(128)
    } else {
      s = -INFINITY;
    }
    sc[d] = s;
  }
  __syncthreads();
  if (d == 0) {
    float m = -INFINITY;
    for (int j = 0; j < 64; ++j) m = fmaxf(m, sc[j]);
    float sum = 0.f;
    for (int j = 0; j < 64; ++j) { float e = __expf(sc[j] - m); sc[j] = e; sum += e; }
    inv = 1.f / sum;
  }
  __syncthreads();
  float o = 0.f;
  for (int kk = 0; kk <= qi; ++kk)
    o += sc[kk] * bf2f(v[(size_t)(b * 64 + kk) * DMODEL + h * HS + d]);
  out[(size_t)t * DMODEL + h * HS + d] = f2bf(o * inv);
}

// h += pos_emb (broadcast over batch)
__global__ __launch_bounds__(256) void k_pos_add(
    float* __restrict__ h, const float* __restrict__ pos) {
  int idx = blockIdx.x * 256 + threadIdx.x;
  int t = idx >> 10, d = idx & 1023;
  h[idx] += pos[(t & 63) * DMODEL + d];
}

// ---------------------------------------------------------------------------
// NoisyTopK router per token: logits, softplus noise, top-2, masked softmax.
// ---------------------------------------------------------------------------
__global__ __launch_bounds__(32) void k_router(
    const unsigned short* __restrict__ xln, const float* __restrict__ Wg,
    const float* __restrict__ bg, const float* __restrict__ Wn,
    const float* __restrict__ bn, const float* __restrict__ noise,
    float* __restrict__ gates) {
  int t = blockIdx.x, e = threadIdx.x;
  __shared__ float noisy[NEXP];
  if (e < NEXP) {
    float lg = bg[e], ln_ = bn[e];
    const unsigned short* xr = xln + (size_t)t * DMODEL;
    for (int d = 0; d < DMODEL; ++d) {
      float xv = bf2f(xr[d]);
      lg  += xv * Wg[d * NEXP + e];
      ln_ += xv * Wn[d * NEXP + e];
    }
    float sp = log1pf(__expf(ln_));  // softplus
    noisy[e] = lg + noise[(size_t)t * NEXP + e] * sp;
  }
  __syncthreads();
  if (e == 0) {
    int i1 = 0, i2 = -1;
    float v1 = noisy[0], v2 = -INFINITY;
    for (int j = 1; j < NEXP; ++j) {
      float nv = noisy[j];
      if (nv > v1) { v2 = v1; i2 = i1; v1 = nv; i1 = j; }
      else if (nv > v2) { v2 = nv; i2 = j; }
    }
    float e2 = __expf(v2 - v1);
    float denom = 1.f + e2;
    for (int j = 0; j < NEXP; ++j) gates[(size_t)t * NEXP + j] = 0.f;
    gates[(size_t)t * NEXP + i1] = 1.f / denom;
    gates[(size_t)t * NEXP + i2] = e2 / denom;
  }
}

// final[t,d] = sum_e eo[e,t,d]
__global__ __launch_bounds__(256) void k_sum_experts(
    const float* __restrict__ eo, float* __restrict__ outp) {
  size_t idx = (size_t)blockIdx.x * 256 + threadIdx.x;
  float s = 0.f;
  #pragma unroll
  for (int e = 0; e < NEXP; ++e) s += eo[(size_t)e * (T_TOK * DMODEL) + idx];
  outp[idx] = s;
}

// feat[b,d] = mean_n second_out[b,n,d]
__global__ __launch_bounds__(256) void k_meanpool(
    const float* __restrict__ so, float* __restrict__ feat) {
  int idx = blockIdx.x * 256 + threadIdx.x;  // b*1024 + d
  int b = idx >> 10, d = idx & 1023;
  float s = 0.f;
  for (int n = 0; n < 64; ++n) s += so[(size_t)(b * 64 + n) * DMODEL + d];
  feat[idx] = s * (1.f / 64.f);
}

// cls[b,j] = feat[b,:] @ Wc[:,j] + bc[j]
__global__ __launch_bounds__(160) void k_classifier(
    const float* __restrict__ feat, const float* __restrict__ Wc,
    const float* __restrict__ bc, float* __restrict__ cls) {
  int tid = threadIdx.x;
  int b = tid / 10, j = tid % 10;
  float acc = bc[j];
  for (int d = 0; d < DMODEL; ++d) acc += feat[b * DMODEL + d] * Wc[d * 10 + j];
  cls[tid] = acc;
}

// ===========================================================================
extern "C" void kernel_launch(void* const* d_in, const int* in_sizes, int n_in,
                              void* d_out, int out_size, void* d_ws, size_t ws_size,
                              hipStream_t stream) {
  (void)in_sizes; (void)n_in; (void)out_size; (void)ws_size;
  const float* x      = (const float*)d_in[0];
  const float* noise1 = (const float*)d_in[1];
  const float* noise2 = (const float*)d_in[2];
  const float* Wpe = (const float*)d_in[3];  const float* bpe = (const float*)d_in[4];
  const float* pos = (const float*)d_in[5];
  const float* g1 = (const float*)d_in[6];   const float* be1 = (const float*)d_in[7];
  const float* g2 = (const float*)d_in[8];   const float* be2 = (const float*)d_in[9];
  const float* g3 = (const float*)d_in[10];  const float* be3 = (const float*)d_in[11];
  const float* Wq = (const float*)d_in[12];  const float* Wk = (const float*)d_in[13];
  const float* Wv = (const float*)d_in[14];  const float* Wp = (const float*)d_in[15];
  const float* bp = (const float*)d_in[16];
  const float* Wg1 = (const float*)d_in[17]; const float* bg1 = (const float*)d_in[18];
  const float* Wn1 = (const float*)d_in[19]; const float* bn1 = (const float*)d_in[20];
  const float* We1a = (const float*)d_in[21]; const float* be1a = (const float*)d_in[22];
  const float* We1b = (const float*)d_in[23]; const float* be1b = (const float*)d_in[24];
  const float* Wg2 = (const float*)d_in[25]; const float* bg2 = (const float*)d_in[26];
  const float* Wn2 = (const float*)d_in[27]; const float* bn2 = (const float*)d_in[28];
  const float* We2a = (const float*)d_in[29]; const float* be2a = (const float*)d_in[30];
  const float* We2b = (const float*)d_in[31]; const float* be2b = (const float*)d_in[32];
  const float* Wc = (const float*)d_in[33];  const float* bc = (const float*)d_in[34];

  // ----- d_out layout (flat concat, floats) -----
  float* out = (float*)d_out;
  const size_t OFF_OUT1 = 0;
  const size_t OFF_OUT2 = 1048576;
  const size_t OFF_FEAT = 2097152;
  const size_t OFF_CLS  = 2113536;
  const size_t OFF_EO1  = 2113696;
  const size_t OFF_EO2  = 12599456;
  const size_t OFF_G1   = 23085216;
  const size_t OFF_G2   = 23095456;

  // ----- workspace carve-up (~270 MB) -----
  size_t off = 0;
  auto carve = [&](size_t bytes) { size_t c = off; off += (bytes + 255) & ~(size_t)255; return c; };
  char* ws = (char*)d_ws;
  float* h0 = (float*)(ws + carve((size_t)T_TOK * DMODEL * 4));
  float* h  = (float*)(ws + carve((size_t)T_TOK * DMODEL * 4));
  unsigned short* hln1 = (unsigned short*)(ws + carve((size_t)T_TOK * DMODEL * 2));
  unsigned short* hln2 = (unsigned short*)(ws + carve((size_t)T_TOK * DMODEL * 2));
  unsigned short* hln3 = (unsigned short*)(ws + carve((size_t)T_TOK * DMODEL * 2));
  unsigned short* qb = (unsigned short*)(ws + carve((size_t)T_TOK * DMODEL * 2));
  unsigned short* kb = (unsigned short*)(ws + carve((size_t)T_TOK * DMODEL * 2));
  unsigned short* vb = (unsigned short*)(ws + carve((size_t)T_TOK * DMODEL * 2));
  unsigned short* ao = (unsigned short*)(ws + carve((size_t)T_TOK * DMODEL * 2));
  unsigned short* Hbuf = (unsigned short*)(ws + carve((size_t)NEXP * T_TOK * DFF * 2));
  // transposed bf16 weights
  unsigned short* WqT = (unsigned short*)(ws + carve((size_t)DMODEL * DMODEL * 2));
  unsigned short* WkT = (unsigned short*)(ws + carve((size_t)DMODEL * DMODEL * 2));
  unsigned short* WvT = (unsigned short*)(ws + carve((size_t)DMODEL * DMODEL * 2));
  unsigned short* WpT = (unsigned short*)(ws + carve((size_t)DMODEL * DMODEL * 2));
  unsigned short* WaT = (unsigned short*)(ws + carve((size_t)NEXP * DMODEL * DFF * 2)); // reused MoE1/MoE2
  unsigned short* WbT = (unsigned short*)(ws + carve((size_t)NEXP * DFF * DMODEL * 2)); // reused MoE1/MoE2

  const long long EXP_W = (long long)DMODEL * DFF;    // per-expert weight elems
  const long long EXP_H = (long long)T_TOK * DFF;
  const long long EXP_Y = (long long)T_TOK * DMODEL;

  dim3 blk256(256);
  dim3 gProj(DMODEL / 128, T_TOK / 128, 1);
  dim3 gFF1(DFF / 128, T_TOK / 128, NEXP);
  dim3 gFF2(DMODEL / 128, T_TOK / 128, NEXP);
  dim3 gTqkv(DMODEL / 32, DMODEL / 32, 1);            // [1024,1024]
  dim3 gTa(DFF / 32, DMODEL / 32, NEXP);              // W1: [K=1024,N=4096]
  dim3 gTb(DMODEL / 32, DFF / 32, NEXP);              // W2: [K=4096,N=1024]

  // 0) one-time attention weight prep (bf16 + transpose)
  k_transpose_bf16<<<gTqkv, blk256, 0, stream>>>(Wq, 0, WqT, 0, DMODEL, DMODEL);
  k_transpose_bf16<<<gTqkv, blk256, 0, stream>>>(Wk, 0, WkT, 0, DMODEL, DMODEL);
  k_transpose_bf16<<<gTqkv, blk256, 0, stream>>>(Wv, 0, WvT, 0, DMODEL, DMODEL);
  k_transpose_bf16<<<gTqkv, blk256, 0, stream>>>(Wp, 0, WpT, 0, DMODEL, DMODEL);

  // 1) patchify + embed
  k_patch_embed<<<T_TOK, blk256, 0, stream>>>(x, Wpe, bpe, h0);
  // 2) ln1 -> bf16
  k_layernorm<<<T_TOK, blk256, 0, stream>>>(h0, g1, be1, hln1);
  // 3) Q/K/V projections (bf16 outputs)
  k_wmma_gemm<F_OUTB><<<gProj, blk256, 0, stream>>>(hln1, 0, WqT, 0, nullptr, 0,
      nullptr, 0, qb, 0, nullptr, nullptr, 0, 0, T_TOK, DMODEL, DMODEL);
  k_wmma_gemm<F_OUTB><<<gProj, blk256, 0, stream>>>(hln1, 0, WkT, 0, nullptr, 0,
      nullptr, 0, kb, 0, nullptr, nullptr, 0, 0, T_TOK, DMODEL, DMODEL);
  k_wmma_gemm<F_OUTB><<<gProj, blk256, 0, stream>>>(hln1, 0, WvT, 0, nullptr, 0,
      nullptr, 0, vb, 0, nullptr, nullptr, 0, 0, T_TOK, DMODEL, DMODEL);
  // 4) attention core (causal softmax + AV)
  k_attention<<<dim3(64, NH, 16), dim3(HS), 0, stream>>>(qb, kb, vb, ao);
  // 5) output projection + residual:  h = h0 + ao @ Wp + bp
  k_wmma_gemm<F_BIAS | F_RES | F_OUTF><<<gProj, blk256, 0, stream>>>(ao, 0, WpT, 0,
      bp, 0, h, 0, nullptr, 0, h0, nullptr, 0, 0, T_TOK, DMODEL, DMODEL);
  // 6) h += pos_emb
  k_pos_add<<<(T_TOK * DMODEL) / 256, blk256, 0, stream>>>(h, pos);
  // 7) ln2 / ln3 -> bf16
  k_layernorm<<<T_TOK, blk256, 0, stream>>>(h, g2, be2, hln2);
  k_layernorm<<<T_TOK, blk256, 0, stream>>>(h, g3, be3, hln3);
  // 8) routers -> gates straight into d_out
  k_router<<<T_TOK, dim3(32), 0, stream>>>(hln2, Wg1, bg1, Wn1, bn1, noise1, out + OFF_G1);
  k_router<<<T_TOK, dim3(32), 0, stream>>>(hln3, Wg2, bg2, Wn2, bn2, noise2, out + OFF_G2);

  // 9) MoE 1: convert weights, then
  //    H = relu(xln2 @ W1[e] + b1[e]);  eo1[e] = (H @ W2[e] + b2[e]) * g[:,e]
  k_transpose_bf16<<<gTa, blk256, 0, stream>>>(We1a, EXP_W, WaT, EXP_W, DMODEL, DFF);
  k_transpose_bf16<<<gTb, blk256, 0, stream>>>(We1b, EXP_W, WbT, EXP_W, DFF, DMODEL);
  k_wmma_gemm<F_BIAS | F_RELU | F_OUTB><<<gFF1, blk256, 0, stream>>>(hln2, 0,
      WaT, EXP_W, be1a, DFF, nullptr, 0, Hbuf, EXP_H, nullptr, nullptr, 0, 0,
      T_TOK, DFF, DMODEL);
  k_wmma_gemm<F_BIAS | F_SCALE | F_OUTF><<<gFF2, blk256, 0, stream>>>(Hbuf, EXP_H,
      WbT, EXP_W, be1b, DMODEL, out + OFF_EO1, EXP_Y, nullptr, 0, nullptr,
      out + OFF_G1, NEXP, 1, T_TOK, DMODEL, DFF);
  k_sum_experts<<<(T_TOK * DMODEL) / 256, blk256, 0, stream>>>(out + OFF_EO1, out + OFF_OUT1);

  // 10) MoE 2 (reuses WaT/WbT/Hbuf; stream-ordered so no hazard)
  k_transpose_bf16<<<gTa, blk256, 0, stream>>>(We2a, EXP_W, WaT, EXP_W, DMODEL, DFF);
  k_transpose_bf16<<<gTb, blk256, 0, stream>>>(We2b, EXP_W, WbT, EXP_W, DFF, DMODEL);
  k_wmma_gemm<F_BIAS | F_RELU | F_OUTB><<<gFF1, blk256, 0, stream>>>(hln3, 0,
      WaT, EXP_W, be2a, DFF, nullptr, 0, Hbuf, EXP_H, nullptr, nullptr, 0, 0,
      T_TOK, DFF, DMODEL);
  k_wmma_gemm<F_BIAS | F_SCALE | F_OUTF><<<gFF2, blk256, 0, stream>>>(Hbuf, EXP_H,
      WbT, EXP_W, be2b, DMODEL, out + OFF_EO2, EXP_Y, nullptr, 0, nullptr,
      out + OFF_G2, NEXP, 1, T_TOK, DMODEL, DFF);
  k_sum_experts<<<(T_TOK * DMODEL) / 256, blk256, 0, stream>>>(out + OFF_EO2, out + OFF_OUT2);

  // 11) mean pool + classifier
  k_meanpool<<<(16 * DMODEL) / 256, blk256, 0, stream>>>(out + OFF_OUT2, out + OFF_FEAT);
  k_classifier<<<1, dim3(160), 0, stream>>>(out + OFF_FEAT, Wc, bc, out + OFF_CLS);
}
